// StackLSTM_21251498181295
// MI455X (gfx1250) — matrix operand order repaired and checked by reading, not compiled, not run on hardware
//
#include <hip/hip_runtime.h>
#include <cstddef>

// ---------------------------------------------------------------------------
// Problem constants
// ---------------------------------------------------------------------------
#define T_LEN 512
#define B_N   64
#define D_IN  512    // input width (layer0 D, layer1 HD) -- both 512
#define G_N   2048   // 4*CD2 gate width
#define HD2_N 256
#define CD2_N 512
#define L_N   2

typedef __attribute__((ext_vector_type(16))) __bf16 v16bf;
typedef __attribute__((ext_vector_type(8)))  float  v8f;

// ---------------------------------------------------------------------------
// Helpers
// ---------------------------------------------------------------------------
__device__ __forceinline__ unsigned short f2bf(float f) {
    unsigned int u = __float_as_uint(f);
    unsigned int r = u + 0x7FFFu + ((u >> 16) & 1u);   // round-to-nearest-even
    return (unsigned short)(r >> 16);
}
__device__ __forceinline__ float bf2f(unsigned short h) {
    return __uint_as_float(((unsigned int)h) << 16);
}
__device__ __forceinline__ float sigmf(float x) {
    return 1.0f / (1.0f + __expf(-x));
}
__device__ __forceinline__ float tanh_fast(float x) {
    float e = __expf(2.0f * x);
    return 1.0f - 2.0f / (e + 1.0f);
}

// A-fragment (16-bit, 16x32): lane l covers row m=l&15, kb=(l<16)?0:8,
//   vector element e <-> k = kb + (e&7) + ((e&8)?16:0)
// B-fragment (16-bit, 32x16): lane l covers col n=l&15, kb2=(l<16)?0:8,
//   vector element e <-> k = kb2 + (e&7) + ((e&8)?16:0)
// C/D-fragment (f32 16x16): lane l covers col n=l&15, element e <-> row e+((l<16)?0:8)

// ---------------------------------------------------------------------------
// Kernel: pack Wi/Wh/Wp (f32) into bf16 B-fragment tile layout.
// WiB: [l][d][gt=128][kt=16][lane=32][16]   (2 MB per (l,d))
// WhB: [l][d][gt=128][kt= 8][lane=32][16]   (1 MB per (l,d))
// WpB: [l][d][nt= 16][kt=16][lane=32][16]   (256 KB per (l,d))
// ---------------------------------------------------------------------------
__global__ void pack_w(const float* __restrict__ Wi, const float* __restrict__ Wh,
                       const float* __restrict__ Wp,
                       unsigned short* __restrict__ WiB,
                       unsigned short* __restrict__ WhB,
                       unsigned short* __restrict__ WpB) {
    int wid = blockIdx.x * 8 + (threadIdx.x >> 5);
    int ln  = threadIdx.x & 31;
    int n   = ln & 15;
    int kb2 = (ln < 16) ? 0 : 8;
    if (wid < 8192) {                       // WiB: 2*2*128*16 tiles
        int l = wid >> 12, d = (wid >> 11) & 1, gt = (wid >> 4) & 127, kt = wid & 15;
        const float* src = Wi + ((size_t)(l * 2 + d) * G_N + gt * 16 + n) * D_IN + kt * 32 + kb2;
        unsigned short* dst = WiB + ((((size_t)(l * 2 + d) * 128 + gt) * 16 + kt) * 32 + ln) * 16;
#pragma unroll
        for (int e = 0; e < 16; ++e) dst[e] = f2bf(src[e < 8 ? e : e + 8]);
    } else if (wid < 12288) {               // WhB: 2*2*128*8 tiles
        int w2 = wid - 8192;
        int l = w2 >> 11, d = (w2 >> 10) & 1, gt = (w2 >> 3) & 127, kt = w2 & 7;
        const float* src = Wh + ((size_t)(l * 2 + d) * G_N + gt * 16 + n) * HD2_N + kt * 32 + kb2;
        unsigned short* dst = WhB + ((((size_t)(l * 2 + d) * 128 + gt) * 8 + kt) * 32 + ln) * 16;
#pragma unroll
        for (int e = 0; e < 16; ++e) dst[e] = f2bf(src[e < 8 ? e : e + 8]);
    } else if (wid < 13312) {               // WpB: 2*2*16*16 tiles
        int w3 = wid - 12288;
        int l = w3 >> 9, d = (w3 >> 8) & 1, nt = (w3 >> 4) & 15, kt = w3 & 15;
        const float* src = Wp + ((size_t)(l * 2 + d) * HD2_N + nt * 16 + n) * CD2_N + kt * 32 + kb2;
        unsigned short* dst = WpB + ((((size_t)(l * 2 + d) * 16 + nt) * 16 + kt) * 32 + ln) * 16;
#pragma unroll
        for (int e = 0; e < 16; ++e) dst[e] = f2bf(src[e < 8 ? e : e + 8]);
    }
}

// ---------------------------------------------------------------------------
// Kernel: convert x (f32 [T*B, 512]) into bf16 A-fragment layout
// xA: [mtG=2048][kt=16][lane=32][16]  (32 MB)
// ---------------------------------------------------------------------------
__global__ void conv_x(const float* __restrict__ x, unsigned short* __restrict__ xA) {
    int tile = blockIdx.x * 8 + (threadIdx.x >> 5);   // 32768 tiles = mtG*16 + kt
    int ln   = threadIdx.x & 31;
    int mtG  = tile >> 4, kt = tile & 15;
    int m    = ln & 15;
    int kb   = (ln < 16) ? 0 : 8;
    const float* src = x + (size_t)(mtG * 16 + m) * D_IN + kt * 32 + kb;
    unsigned short* dst = xA + ((size_t)tile * 32 + ln) * 16;
#pragma unroll
    for (int e = 0; e < 8; ++e) dst[e] = f2bf(src[e]);          // k = kb..kb+7
#pragma unroll
    for (int e = 0; e < 8; ++e) dst[8 + e] = f2bf(src[16 + e]); // k = 16+kb..
}

// ---------------------------------------------------------------------------
// Kernel: pre = x @ Wi.T for both directions of one layer.
// preC: [d=2][mtG=2048][gt=128][lane=32][8]  f32 C-fragment layout (512 MB)
// One wave computes 16(M) x 64(N); software-pipelined over K (prefetch kt+1).
// ---------------------------------------------------------------------------
__global__ void gemm_pre(const unsigned short* __restrict__ xA,
                         const unsigned short* __restrict__ WiBl,  // layer base
                         float* __restrict__ preC) {
    int wid = blockIdx.x * 8 + (threadIdx.x >> 5);
    if (wid >= 131072) return;
    int ln  = threadIdx.x & 31;
    int d   = wid >> 16;
    int rem = wid & 65535;
    int mtG = rem & 2047;
    int gt0 = (rem >> 11) * 4;
    const unsigned short* WiBd = WiBl + (size_t)d * 128 * 16 * 32 * 16;
    const unsigned short* aP = xA + ((size_t)mtG * 16 * 32 + ln) * 16;   // + kt*512

    v8f C[4];
    v8f z8 = {0.f, 0.f, 0.f, 0.f, 0.f, 0.f, 0.f, 0.f};
    C[0] = z8; C[1] = z8; C[2] = z8; C[3] = z8;

    v16bf Ac = *(const v16bf*)(aP);
    v16bf Bc[4];
#pragma unroll
    for (int j = 0; j < 4; ++j)
        Bc[j] = *(const v16bf*)(WiBd + (((size_t)(gt0 + j) * 16 + 0) * 32 + ln) * 16);

#pragma unroll
    for (int kt = 0; kt < 16; ++kt) {
        v16bf An = Ac;
        v16bf Bn[4] = {Bc[0], Bc[1], Bc[2], Bc[3]};
        if (kt < 15) {                                   // prefetch next K step
            An = *(const v16bf*)(aP + (size_t)(kt + 1) * 512);
#pragma unroll
            for (int j = 0; j < 4; ++j)
                Bn[j] = *(const v16bf*)(WiBd + (((size_t)(gt0 + j) * 16 + kt + 1) * 32 + ln) * 16);
        }
#pragma unroll
        for (int j = 0; j < 4; ++j)
            C[j] = __builtin_amdgcn_wmma_f32_16x16x32_bf16(false, Ac, false, Bc[j],
                                                           (short)0, C[j], false, false);
        Ac = An;
#pragma unroll
        for (int j = 0; j < 4; ++j) Bc[j] = Bn[j];
    }
    float* outb = preC + ((((size_t)d * 2048 + mtG) * 128 + gt0) * 32 + ln) * 8;
#pragma unroll
    for (int j = 0; j < 4; ++j) *(v8f*)(outb + (size_t)j * 256) = C[j];
}

// ---------------------------------------------------------------------------
// Recurrence kernel: one workgroup (512 thr = 16 waves) per direction.
// LDS (232 KB of 320 KB/WGP):
//   c_lds : C-frag f32   [mb=4][jt=32][lane=32][8]   128 KB
//   sA    : A-frag bf16  [mb=4][kt=16][lane=32][16]   64 KB
//   hA    : A-frag bf16  [mb=4][kt= 8][lane=32][16]   32 KB
//   bias_s: f32 [2048]                                  8 KB
// ---------------------------------------------------------------------------
__global__ __launch_bounds__(512) void lstm_rec(
    const float* __restrict__ preC,
    const unsigned short* __restrict__ WhBl,  // layer base
    const unsigned short* __restrict__ WpBl,  // layer base
    const float* __restrict__ bias,           // layer base [2][2048]
    const int* __restrict__ lens,
    float* __restrict__ outp,                 // [T,B,512] f32 or nullptr
    unsigned short* __restrict__ xA,          // next-layer A-frag buffer
    int write_xA,
    float* __restrict__ state) {              // h(64*512) then c(64*1024), or nullptr

    __shared__ __attribute__((aligned(32))) float          c_lds[4 * 32 * 32 * 8];
    __shared__ __attribute__((aligned(32))) unsigned short sA[4 * 16 * 32 * 16];
    __shared__ __attribute__((aligned(32))) unsigned short hA[4 * 8 * 32 * 16];
    __shared__ float bias_s[G_N];
    __shared__ int lens_s[B_N];

    const int tid = threadIdx.x;
    const int d   = blockIdx.x;

    const float* bd = bias + (size_t)d * G_N;
    for (int i = tid; i < 4 * 32 * 32 * 8; i += 512) c_lds[i] = 0.f;
    for (int i = tid; i < 4 * 8 * 32 * 16; i += 512) hA[i] = 0;
    for (int i = tid; i < G_N; i += 512) bias_s[i] = bd[i];
    if (tid < B_N) lens_s[tid] = lens[tid];
    __syncthreads();

    const int wv = tid >> 5, ln = tid & 31;
    const int n = ln & 15;
    const int rowoff = (ln < 16) ? 0 : 8;
    const int mb = wv >> 2;        // 0..3 : batch tile (both stages)
    const int jb = wv & 3;         // 0..3 : j-tile group (stage 1)
    const int ng = wv & 3;         // 0..3 : nt group (stage 3)

    const unsigned short* WhBd = WhBl + (size_t)d * 128 * 8 * 32 * 16;
    const unsigned short* WpBd = WpBl + (size_t)d * 16 * 16 * 32 * 16;
    v8f z8 = {0.f, 0.f, 0.f, 0.f, 0.f, 0.f, 0.f, 0.f};

    for (int s = 0; s < T_LEN; ++s) {
        const int ts = d ? (T_LEN - 1 - s) : s;

        // Prefetch next step's preC slice (only HBM stream in the loop).
        if (s + 1 < T_LEN) {
            const int tsn = d ? (ts - 1) : (ts + 1);
            const float* pn = preC + (((size_t)d * 2048 + (size_t)(4 * tsn + mb)) * 128) * 256;
#pragma unroll
            for (int a = 0; a < 8; ++a) {
#pragma unroll
                for (int q = 0; q < 4; ++q) {
                    const int gt = q * 32 + (jb + 4 * a);
                    __builtin_prefetch((const void*)(pn + ((size_t)gt * 32 + ln) * 8), 0, 3);
                }
            }
        }

        // ---- Stage 1: gates + cell update -------------------------------
        v16bf A[8];
#pragma unroll
        for (int kt = 0; kt < 8; ++kt)
            A[kt] = *(const v16bf*)(hA + ((((mb * 8 + kt) * 32) + ln) << 4));

        const float* pc = preC + (((size_t)d * 2048 + (size_t)(4 * ts + mb)) * 128) * 256;

        for (int a = 0; a < 8; ++a) {
            const int jt = jb + 4 * a;
            v8f C[4];
#pragma unroll
            for (int q = 0; q < 4; ++q)
                C[q] = *(const v8f*)(pc + ((size_t)(q * 32 + jt) * 32 + ln) * 8);

            // Software-pipelined gate GEMM; q varies fastest so 4 consecutive
            // WMMAs share A[kt] and the accumulator rotates (no serial C chain).
            v16bf Bc = *(const v16bf*)(WhBd + (((size_t)jt * 8) * 32 + ln) * 16);
#pragma unroll
            for (int idx = 0; idx < 32; ++idx) {
                const int kt = idx >> 2, q = idx & 3;
                v16bf Bn = Bc;
                if (idx < 31) {
                    const int i2 = idx + 1;
                    const int gt2 = (i2 & 3) * 32 + jt;
                    Bn = *(const v16bf*)(WhBd + (((size_t)gt2 * 8 + (i2 >> 2)) * 32 + ln) * 16);
                }
                C[q] = __builtin_amdgcn_wmma_f32_16x16x32_bf16(false, A[kt], false, Bc,
                                                               (short)0, C[q], false, false);
                Bc = Bn;
            }

            const int j0 = jt * 16 + n;                       // gate column 0..511
            const float bi = bias_s[j0], bf_ = bias_s[512 + j0];
            const float bg = bias_s[1024 + j0], bo = bias_s[1536 + j0];
            const int cbase = ((mb * 32 + jt) * 32 + ln) * 8;
            const int ktS = j0 >> 5, kmod = j0 & 31;
            const int l2b = (kmod & 8) ? 16 : 0;
            const int half = ((kmod & 16) ? 8 : 0) + (kmod & 7);
#pragma unroll
            for (int e = 0; e < 8; ++e) {
                const int rowt = rowoff + e;
                const int brow = mb * 16 + rowt;
                const bool m = ts < lens_s[brow];
                const float co = c_lds[cbase + e];
                const float iv = sigmf(C[0][e] + bi);
                const float fv = sigmf(C[1][e] + bf_);
                const float gv = tanh_fast(C[2][e] + bg);
                const float ov = sigmf(C[3][e] + bo);
                const float cn = fv * co + iv * gv;
                const float sv = ov * tanh_fast(cn);
                c_lds[cbase + e] = m ? cn : co;
                sA[(((mb * 16 + ktS) * 32 + (rowt + l2b)) << 4) + half] = f2bf(sv);
            }
        }
        __syncthreads();

        // ---- Stage 3: projection + masked writeback ---------------------
        {
            v8f H[4];
            H[0] = z8; H[1] = z8; H[2] = z8; H[3] = z8;

            v16bf Ac = *(const v16bf*)(sA + (((mb * 16) * 32 + ln) << 4));
            v16bf Bc[4];
#pragma unroll
            for (int e2 = 0; e2 < 4; ++e2)
                Bc[e2] = *(const v16bf*)(WpBd + ((((size_t)(ng * 4 + e2) * 16) * 32 + ln) << 4));

#pragma unroll
            for (int kt = 0; kt < 16; ++kt) {
                v16bf An = Ac;
                v16bf Bn[4] = {Bc[0], Bc[1], Bc[2], Bc[3]};
                if (kt < 15) {                             // prefetch next K step
                    An = *(const v16bf*)(sA + (((mb * 16 + kt + 1) * 32 + ln) << 4));
#pragma unroll
                    for (int e2 = 0; e2 < 4; ++e2)
                        Bn[e2] = *(const v16bf*)(WpBd +
                                 ((((size_t)(ng * 4 + e2) * 16 + kt + 1) * 32 + ln) << 4));
                }
#pragma unroll
                for (int e2 = 0; e2 < 4; ++e2)
                    H[e2] = __builtin_amdgcn_wmma_f32_16x16x32_bf16(false, Ac, false, Bc[e2],
                                                                    (short)0, H[e2], false, false);
                Ac = An;
#pragma unroll
                for (int e2 = 0; e2 < 4; ++e2) Bc[e2] = Bn[e2];
            }

#pragma unroll
            for (int e2 = 0; e2 < 4; ++e2) {
                const int nt = ng * 4 + e2;
                const int p = nt * 16 + n;                    // h column 0..255
                const int ktH = p >> 5, kmod = p & 31;
                const int l2b = (kmod & 8) ? 16 : 0;
                const int half = ((kmod & 16) ? 8 : 0) + (kmod & 7);
                const int xcol = d * 256 + p;
                const int ktX = xcol >> 5, kmX = xcol & 31;
                const int l2bx = (kmX & 8) ? 16 : 0;
                const int halfX = ((kmX & 16) ? 8 : 0) + (kmX & 7);
#pragma unroll
                for (int e = 0; e < 8; ++e) {
                    const int rowt = rowoff + e;
                    const int brow = mb * 16 + rowt;
                    const bool m = ts < lens_s[brow];
                    const float hn = H[e2][e];
                    const int hidx = (((mb * 8 + ktH) * 32 + (rowt + l2b)) << 4) + half;
                    const unsigned short hold = hA[hidx];
                    const unsigned short hnb = f2bf(hn);
                    hA[hidx] = m ? hnb : hold;                // h2 = m ? h_new : h
                    if (outp)
                        outp[((size_t)ts * B_N + brow) * 512 + d * 256 + p] = m ? hn : 0.f;
                    if (write_xA) {
                        const unsigned short xv = m ? hnb : (unsigned short)0;  // out_t
                        xA[((((size_t)(ts * 4 + mb) * 16 + ktX) * 32 + (rowt + l2bx)) << 4) + halfX] = xv;
                    }
                }
            }
        }
        __syncthreads();
    }

    // ---- Final (h, c) state for the last layer --------------------------
    if (state) {
        for (int i = tid; i < B_N * HD2_N; i += 512) {
            const int b = i >> 8, p = i & 255;
            const int mbb = b >> 4, rowt = b & 15;
            const int ktH = p >> 5, kmod = p & 31;
            const int l2b = (kmod & 8) ? 16 : 0;
            const int half = ((kmod & 16) ? 8 : 0) + (kmod & 7);
            const unsigned short hv = hA[(((mbb * 8 + ktH) * 32 + (rowt + l2b)) << 4) + half];
            state[(size_t)b * 512 + d * 256 + p] = bf2f(hv);
        }
        float* cdst = state + (size_t)B_N * 512;
        for (int i = tid; i < B_N * CD2_N; i += 512) {
            const int b = i >> 9, j = i & 511;
            const int mbb = b >> 4, rowt = b & 15;
            const int jt = j >> 4, nn = j & 15;
            const int lane2 = nn + ((rowt & 8) ? 16 : 0);
            const int e = rowt & 7;
            cdst[(size_t)b * 1024 + d * 512 + j] =
                c_lds[((mbb * 32 + jt) * 32 + lane2) * 8 + e];
        }
    }
}

// ---------------------------------------------------------------------------
// Host launcher
// ---------------------------------------------------------------------------
extern "C" void kernel_launch(void* const* d_in, const int* in_sizes, int n_in,
                              void* d_out, int out_size, void* d_ws, size_t ws_size,
                              hipStream_t stream) {
    const float* x    = (const float*)d_in[0];
    const int*   lens = (const int*)d_in[1];
    const float* Wi   = (const float*)d_in[2];
    const float* Wh   = (const float*)d_in[3];
    const float* bv   = (const float*)d_in[4];
    const float* Wp   = (const float*)d_in[5];
    float* out = (float*)d_out;

    char* ws = (char*)d_ws;
    const size_t SZ_XA  = (size_t)2048 * 16 * 32 * 32;                 // 32 MB (bytes)
    const size_t SZ_WIB = (size_t)2 * 2 * 128 * 16 * 32 * 32;          // 8 MB
    const size_t SZ_WHB = (size_t)2 * 2 * 128 * 8 * 32 * 32;           // 4 MB
    const size_t SZ_WPB = (size_t)2 * 2 * 16 * 16 * 32 * 32;           // 1 MB
    unsigned short* xA  = (unsigned short*)ws;
    unsigned short* WiB = (unsigned short*)(ws + SZ_XA);
    unsigned short* WhB = (unsigned short*)(ws + SZ_XA + SZ_WIB);
    unsigned short* WpB = (unsigned short*)(ws + SZ_XA + SZ_WIB + SZ_WHB);
    float* preC = (float*)(ws + SZ_XA + SZ_WIB + SZ_WHB + SZ_WPB);     // 512 MB

    const size_t H_OFF = (size_t)T_LEN * B_N * 512;                    // 16777216

    pack_w<<<1664, 256, 0, stream>>>(Wi, Wh, Wp, WiB, WhB, WpB);
    conv_x<<<4096, 256, 0, stream>>>(x, xA);

    for (int l = 0; l < L_N; ++l) {
        const unsigned short* WiBl = WiB + (size_t)l * 2 * 128 * 16 * 32 * 16;
        const unsigned short* WhBl = WhB + (size_t)l * 2 * 128 * 8 * 32 * 16;
        const unsigned short* WpBl = WpB + (size_t)l * 2 * 16 * 16 * 32 * 16;
        const float* bl = bv + (size_t)l * 2 * G_N;

        gemm_pre<<<16384, 256, 0, stream>>>(xA, WiBl, preC);
        lstm_rec<<<2, 512, 0, stream>>>(preC, WhBl, WpBl, bl, lens,
                                        (l == L_N - 1) ? out : nullptr,
                                        xA, (l == 0) ? 1 : 0,
                                        (l == L_N - 1) ? (out + H_OFF) : nullptr);
    }
}
